// LSTMDynamics_49950469653088
// MI455X (gfx1250) — matrix operand order repaired and checked by reading, not compiled, not run on hardware
//
#include <hip/hip_runtime.h>
#include <hip/hip_bf16.h>

// ---------------------------------------------------------------------------
// LSTM dynamics rollout for MI455X (gfx1250, wave32, WMMA).
//
//  * grid = B/16 workgroups; each WG owns 16 batch rows for the whole rollout.
//  * 8 waves/WG split the 128 hidden units: wave w owns gate columns
//    [16w,16w+16) of each of the 4 gate blocks (i,f,g,o).
//  * Weight placement:
//      - W0 (20 frags = 160 VGPRs) + W1 k-tiles 0..3 (16 frags = 128 VGPRs)
//        register-resident per wave.
//      - W1 k-tiles 4..7 staged in LDS (128 KB dynamic smem) and read as
//        32B/lane fragments each step.
//  * Activations are f16 A-matrices in LDS:  A0 = [x | h0] (16x160),
//    A1 = [h0n | h1] (16x256).  x kept in f32 in LDS for drift-free Euler.
//  * Cell state c0/c1 in accumulator-layout VGPRs; bias preloaded into C.
//  * v_wmma_f32_16x16x32_f16, f32 accumulate.
//  * Gate nonlinearities via hardware V_TANH_F32 (branchless, co-executes
//    with WMMA); A fragments double-buffered to hide DS latency.
// ---------------------------------------------------------------------------

typedef __attribute__((ext_vector_type(16))) _Float16 v16h;
typedef __attribute__((ext_vector_type(8)))  _Float16 v8h;
typedef __attribute__((ext_vector_type(8)))  float    v8f;

#define HID    128
#define SDIM   32
#define ROWS   16      // batch rows per workgroup
#define NWAVE  8
#define KT0    5       // K = 160 -> 5 k-tiles of 32
#define KT1    8       // K = 256 -> 8 k-tiles of 32
#define KT1R   4       // W1 k-tiles kept in registers
#define A0_STR 168     // halfs; 84 dwords/row staggers LDS banks
#define A1_STR 264     // halfs; 132 dwords/row

static __device__ __forceinline__ v8f wmma16(v16h a, v16h b, v8f c) {
  // (neg_a, A, neg_b, B, c_mod, C, reuse_a, reuse_b)
  return __builtin_amdgcn_wmma_f32_16x16x32_f16(false, a, false, b, (short)0, c,
                                                false, false);
}

static __device__ __forceinline__ v8f splat8(float x) {
  v8f v = {x, x, x, x, x, x, x, x};
  return v;
}

// Hardware transcendental tanh (CDNA5 TRANS op).  Trailing V_NOP satisfies
// the ISA 7.4 rule: one independent op/V_NOP after a TRANS op before its
// output may be consumed (hardware does not interlock co-executing
// multicycle ops from the same wave).
static __device__ __forceinline__ float tanh_hw(float x) {
  float r;
  asm volatile("v_tanh_f32 %0, %1\n\tv_nop" : "=v"(r) : "v"(x));
  return r;
}

static __device__ __forceinline__ float sigmoid_hw(float x) {
  return fmaf(0.5f, tanh_hw(0.5f * x), 0.5f);
}

// 16-bit A-matrix 16x32 fragment from row-major f16 LDS (ISA 7.12.2 layout):
// lane<16 : M=lane,    halves 0..7 = K kbase+0..7,   8..15 = K kbase+16..23
// lane>=16: M=lane-16, halves 0..7 = K kbase+8..15,  8..15 = K kbase+24..31
static __device__ __forceinline__ v16h load_afrag(const _Float16* base,
                                                  int stride, int kbase,
                                                  int lane) {
  const int row = lane & 15;
  const int k0  = kbase + ((lane >> 4) << 3);
  const _Float16* p = base + row * stride + k0;
  union { v16h v; v8h h[2]; } u;
  u.h[0] = *(const v8h*)(p);        // 16B -> ds_load_b128
  u.h[1] = *(const v8h*)(p + 16);   // 16B -> ds_load_b128
  return u.v;
}

// ---------------------------------------------------------------------------
// Pack f32 weights into per-wave WMMA-B fragment order (f16).
// B-fragment (32x16, f16): lane<16: N=lane,   halves 0..7 -> K kt*32+0..7,  8..15 -> K kt*32+16..23
//                          lane>=16: N=lane-16, halves 0..7 -> K kt*32+8..15, 8..15 -> K kt*32+24..31
// Fragment storage: 512 halfs, lane-major (lane*16 .. lane*16+15).
// W0 pack order: [wave][kt][gate];  W1 same;  Wout: [ntile][kt].
// ---------------------------------------------------------------------------
__global__ void lstm_pack_weights(const float* __restrict__ W0,
                                  const float* __restrict__ W1,
                                  const float* __restrict__ Wout,
                                  _Float16* __restrict__ w0p,
                                  _Float16* __restrict__ w1p,
                                  _Float16* __restrict__ wop) {
  const int N0 = NWAVE * 4 * KT0 * 512;  // 81920
  const int N1 = NWAVE * 4 * KT1 * 512;  // 131072
  const int NO = 2 * 4 * 512;            // 4096
  int idx = blockIdx.x * blockDim.x + threadIdx.x;
  if (idx < N0) {
    int e = idx & 511, f = idx >> 9;
    int lane = e >> 4, j = e & 15;
    int g = f & 3, kt = (f >> 2) % KT0, w = f / (4 * KT0);
    int K = kt * 32 + ((lane >> 4) << 3) + (j & 7) + ((j >> 3) << 4);
    int n = g * HID + w * 16 + (lane & 15);
    w0p[idx] = (_Float16)W0[K * 4 * HID + n];
  } else if (idx < N0 + N1) {
    int i2 = idx - N0;
    int e = i2 & 511, f = i2 >> 9;
    int lane = e >> 4, j = e & 15;
    int g = f & 3, kt = (f >> 2) % KT1, w = f / (4 * KT1);
    int K = kt * 32 + ((lane >> 4) << 3) + (j & 7) + ((j >> 3) << 4);
    int n = g * HID + w * 16 + (lane & 15);
    w1p[i2] = (_Float16)W1[K * 4 * HID + n];
  } else if (idx < N0 + N1 + NO) {
    int i2 = idx - N0 - N1;
    int e = i2 & 511, f = i2 >> 9;  // f = nt*4 + kt
    int lane = e >> 4, j = e & 15;
    int kt = f & 3, nt = f >> 2;
    int K = kt * 32 + ((lane >> 4) << 3) + (j & 7) + ((j >> 3) << 4);
    int n = nt * 16 + (lane & 15);
    wop[i2] = (_Float16)Wout[K * SDIM + n];
  }
}

// ---------------------------------------------------------------------------
// Main rollout kernel.
// ---------------------------------------------------------------------------
__global__ __launch_bounds__(256, 1)
void lstm_rollout(const float* __restrict__ x0,
                  const float* __restrict__ b0,
                  const float* __restrict__ b1,
                  const float* __restrict__ bout,
                  const float* __restrict__ dts,
                  const _Float16* __restrict__ w0p,
                  const _Float16* __restrict__ w1p,
                  const _Float16* __restrict__ wop,
                  float* __restrict__ traj,
                  int T) {
  __shared__ __attribute__((aligned(16))) _Float16 A0[ROWS * A0_STR];
  __shared__ __attribute__((aligned(16))) _Float16 A1[ROWS * A1_STR];
  __shared__ __attribute__((aligned(16))) float    xs[ROWS * SDIM];
  __shared__ __attribute__((aligned(16))) _Float16 wo[2 * 4 * 512];
  extern __shared__ _Float16 w1hi[];  // 8 waves * 16 frags * 512 halfs = 128KB

  const int tid   = threadIdx.x;
  const int lane  = tid & 31;
  const int wv    = tid >> 5;
  const int brow0 = blockIdx.x * ROWS;

  // ---- init LDS: zero h regions, stage Wout fragments, load x0 ----
  for (int i = tid; i < ROWS * A0_STR; i += 256) A0[i] = (_Float16)0.0f;
  for (int i = tid; i < ROWS * A1_STR; i += 256) A1[i] = (_Float16)0.0f;
  for (int i = tid; i < 2 * 4 * 512; i += 256) wo[i] = wop[i];
  for (int i = tid; i < ROWS * SDIM; i += 256) {
    int r = i >> 5, c = i & 31;
    float v = x0[(brow0 + r) * SDIM + c];
    xs[i] = v;
    A0[r * A0_STR + c] = (_Float16)v;
    traj[(size_t)(brow0 + r) * (size_t)(T + 1) * SDIM + c] = v;  // t = 0 slot
  }

  // ---- stage W1 k-tiles 4..7 into this wave's LDS region (16 KB) ----
  _Float16* w1l = w1hi + wv * ((KT1 - KT1R) * 4 * 512);
  {
    const _Float16* src = w1p + (size_t)wv * (KT1 * 4 * 512) + KT1R * 4 * 512;
    for (int i = lane; i < (KT1 - KT1R) * 4 * 512 / 16; i += 32)
      *(v16h*)(w1l + i * 16) = *(const v16h*)(src + i * 16);
  }

  // ---- register-resident weight fragments (288 VGPRs/wave) ----
  v16h w0f[KT0][4];
  {
    const _Float16* p = w0p + (size_t)wv * (KT0 * 4 * 512) + lane * 16;
#pragma unroll
    for (int kt = 0; kt < KT0; ++kt)
#pragma unroll
      for (int g = 0; g < 4; ++g)
        w0f[kt][g] = *(const v16h*)(p + (kt * 4 + g) * 512);
  }
  v16h w1f[KT1R][4];
  {
    const _Float16* p = w1p + (size_t)wv * (KT1 * 4 * 512) + lane * 16;
#pragma unroll
    for (int kt = 0; kt < KT1R; ++kt)
#pragma unroll
      for (int g = 0; g < 4; ++g)
        w1f[kt][g] = *(const v16h*)(p + (kt * 4 + g) * 512);
  }

  // per-lane gate column owned by this wave: hidden unit index
  const int coll = (wv << 4) + (lane & 15);
  const float bi0 = b0[0 * HID + coll], bf0 = b0[1 * HID + coll];
  const float bg0 = b0[2 * HID + coll], bo0 = b0[3 * HID + coll];
  const float bi1 = b1[0 * HID + coll], bf1 = b1[1 * HID + coll];
  const float bg1 = b1[2 * HID + coll], bo1 = b1[3 * HID + coll];
  float bov = 0.0f, dtv = 0.0f;
  if (wv < 2) {  // waves 0,1 own the 2 output n-tiles (32 state dims)
    bov = bout[coll];
    dtv = dts[coll] * 0.01f;  // dt_scale * DT folded
  }

  v8f c0s = splat8(0.0f);
  v8f c1s = splat8(0.0f);
  const int rbase = (lane >> 4) << 3;  // C/D layout: rows r / 8+r

  __syncthreads();

#pragma unroll 1
  for (int t = 0; t < T; ++t) {
    // ================= cell 0: gates = [x|h0] @ W0 + b0 =================
    v8f ai = splat8(bi0), af = splat8(bf0), ag = splat8(bg0), ao = splat8(bo0);
    {
      v16h nxt = load_afrag(A0, A0_STR, 0, lane);
#pragma unroll
      for (int kt = 0; kt < KT0; ++kt) {
        v16h a = nxt;
        if (kt + 1 < KT0) nxt = load_afrag(A0, A0_STR, (kt + 1) * 32, lane);
        ai = wmma16(a, w0f[kt][0], ai);
        af = wmma16(a, w0f[kt][1], af);
        ag = wmma16(a, w0f[kt][2], ag);
        ao = wmma16(a, w0f[kt][3], ao);
      }
    }
    __syncthreads();  // all A0 reads complete before h0 overwrite
#pragma unroll
    for (int r = 0; r < 8; ++r) {
      float iv = sigmoid_hw(ai[r]);
      float fv = sigmoid_hw(af[r]);
      float gv = tanh_hw(ag[r]);
      float ov = sigmoid_hw(ao[r]);
      float cn = fv * c0s[r] + iv * gv;
      c0s[r] = cn;
      _Float16 hh = (_Float16)(ov * tanh_hw(cn));
      int row = rbase + r;
      A0[row * A0_STR + SDIM + coll] = hh;  // h0 for next step's cell0
      A1[row * A1_STR + coll]        = hh;  // h0n for cell1
    }
    __syncthreads();  // h0n visible

    // ================= cell 1: gates = [h0n|h1] @ W1 + b1 ===============
    v8f zi = splat8(bi1), zf = splat8(bf1), zg = splat8(bg1), zo = splat8(bo1);
    {
      v16h nxt = load_afrag(A1, A1_STR, 0, lane);
#pragma unroll
      for (int kt = 0; kt < KT1; ++kt) {
        v16h a = nxt;
        if (kt + 1 < KT1) nxt = load_afrag(A1, A1_STR, (kt + 1) * 32, lane);
        if (kt < KT1R) {
          zi = wmma16(a, w1f[kt][0], zi);
          zf = wmma16(a, w1f[kt][1], zf);
          zg = wmma16(a, w1f[kt][2], zg);
          zo = wmma16(a, w1f[kt][3], zo);
        } else {
          const _Float16* wb = w1l + ((kt - KT1R) * 4) * 512 + lane * 16;
          zi = wmma16(a, *(const v16h*)(wb + 0 * 512), zi);
          zf = wmma16(a, *(const v16h*)(wb + 1 * 512), zf);
          zg = wmma16(a, *(const v16h*)(wb + 2 * 512), zg);
          zo = wmma16(a, *(const v16h*)(wb + 3 * 512), zo);
        }
      }
    }
    __syncthreads();  // all A1 reads complete before h1 overwrite
#pragma unroll
    for (int r = 0; r < 8; ++r) {
      float iv = sigmoid_hw(zi[r]);
      float fv = sigmoid_hw(zf[r]);
      float gv = tanh_hw(zg[r]);
      float ov = sigmoid_hw(zo[r]);
      float cn = fv * c1s[r] + iv * gv;
      c1s[r] = cn;
      A1[(rbase + r) * A1_STR + HID + coll] = (_Float16)(ov * tanh_hw(cn));
    }
    __syncthreads();  // h1n visible

    // ============ output projection + Euler update (waves 0,1) ==========
    if (wv < 2) {
      v8f dx = splat8(bov);
      v16h nxt = load_afrag(A1, A1_STR, HID, lane);
#pragma unroll
      for (int kt = 0; kt < 4; ++kt) {
        v16h a = nxt;
        if (kt + 1 < 4) nxt = load_afrag(A1, A1_STR, HID + (kt + 1) * 32, lane);
        v16h bw = *(const v16h*)(wo + (wv * 4 + kt) * 512 + lane * 16);
        dx = wmma16(a, bw, dx);
      }
#pragma unroll
      for (int r = 0; r < 8; ++r) {
        int row = rbase + r;
        float xn = xs[row * SDIM + coll] + dx[r] * dtv;
        xs[row * SDIM + coll] = xn;
        A0[row * A0_STR + coll] = (_Float16)xn;
        traj[(size_t)(brow0 + row) * (size_t)(T + 1) * SDIM +
             (size_t)(t + 1) * SDIM + coll] = xn;
      }
    }
    __syncthreads();  // x writes visible before next step's cell0 reads
  }
}

// ---------------------------------------------------------------------------
extern "C" void kernel_launch(void* const* d_in, const int* in_sizes, int n_in,
                              void* d_out, int out_size, void* d_ws,
                              size_t ws_size, hipStream_t stream) {
  const float* x0   = (const float*)d_in[0];
  const float* W0   = (const float*)d_in[1];
  const float* b0   = (const float*)d_in[2];
  const float* W1   = (const float*)d_in[3];
  const float* b1   = (const float*)d_in[4];
  const float* Wout = (const float*)d_in[5];
  const float* bout = (const float*)d_in[6];
  const float* dts  = (const float*)d_in[7];

  const int B = in_sizes[0] / SDIM;                // 8192
  const int T = out_size / (B * SDIM) - 1;         // 512 (robust, host-side)

  _Float16* w0p = (_Float16*)d_ws;                 // 81920 halfs
  _Float16* w1p = w0p + NWAVE * 4 * KT0 * 512;     // 131072 halfs
  _Float16* wop = w1p + NWAVE * 4 * KT1 * 512;     // 4096 halfs

  const int packN = NWAVE * 4 * (KT0 + KT1) * 512 + 2 * 4 * 512;
  lstm_pack_weights<<<(packN + 255) / 256, 256, 0, stream>>>(W0, W1, Wout,
                                                             w0p, w1p, wop);

  const size_t w1hiBytes =
      (size_t)NWAVE * (KT1 - KT1R) * 4 * 512 * sizeof(_Float16);  // 128 KB
  lstm_rollout<<<B / ROWS, 256, w1hiBytes, stream>>>(
      x0, b0, b1, bout, dts, w0p, w1p, wop, (float*)d_out, T);
}
